// HLG_HIMP_80650895884472
// MI455X (gfx1250) — matrix operand order repaired and checked by reading, not compiled, not run on hardware
//
#include <hip/hip_runtime.h>

typedef __attribute__((ext_vector_type(16))) _Float16 v16h;
typedef __attribute__((ext_vector_type(8)))  float    v8f;

#define CDIV(a,b) (((a)+(b)-1)/(b))

struct alignas(8) H4 { _Float16 v[4]; };

// ---------------------------------------------------------------------------
// utility kernels
// ---------------------------------------------------------------------------
__global__ void zero_k(float* __restrict__ p, size_t n) {
  size_t i = (size_t)blockIdx.x * blockDim.x + threadIdx.x;
  if (i < n) p[i] = 0.f;
}

// out[row] = t0[f0] + t1[f1] + t2[f2]   (embedding-sum encoder)
__global__ void encode3_k(const int* __restrict__ feats,
                          const float* __restrict__ t0,
                          const float* __restrict__ t1,
                          const float* __restrict__ t2,
                          float* __restrict__ out, int M) {
  size_t i = (size_t)blockIdx.x * blockDim.x + threadIdx.x;
  if (i >= (size_t)M * 64) return;
  int row = (int)(i >> 6), c = (int)(i & 63);
  int f0 = feats[row * 3 + 0], f1 = feats[row * 3 + 1], f2 = feats[row * 3 + 2];
  out[i] = t0[f0 * 64 + c] + t1[f1 * 64 + c] + t2[f2 * 64 + c];
}

// acc[d] += src[s]; cnt[d] += 1   (segment-sum scatter, optional src gather and
// dest indirection d = dmap[didx[row]])
__global__ void scatter_rows_k(const float* __restrict__ src,
                               const int* __restrict__ sidx,
                               const int* __restrict__ didx,
                               const int* __restrict__ dmap,
                               float* __restrict__ acc,
                               float* __restrict__ cnt, int M) {
  size_t i = (size_t)blockIdx.x * blockDim.x + threadIdx.x;
  if (i >= (size_t)M * 64) return;
  int row = (int)(i >> 6), c = (int)(i & 63);
  int s = sidx ? sidx[row] : row;
  int d = didx[row];
  if (dmap) d = dmap[d];
  atomicAdd(&acc[(size_t)d * 64 + c], src[(size_t)s * 64 + c]);
  if (c == 0) atomicAdd(&cnt[d], 1.0f);
}

// per-column sum and sum-of-squares over M rows into st[0..63], st[64..127]
__global__ void colstats_k(const float* __restrict__ X, int M,
                           float* __restrict__ st) {
  __shared__ float ssum[4][64];
  __shared__ float ssq[4][64];
  int t = threadIdx.x, c = t & 63, gR = t >> 6;
  float s = 0.f, q = 0.f;
  for (int r = blockIdx.x * 4 + gR; r < M; r += gridDim.x * 4) {
    float v = X[(size_t)r * 64 + c];
    s += v; q += v * v;
  }
  ssum[gR][c] = s; ssq[gR][c] = q;
  __syncthreads();
  if (t < 64) {
    atomicAdd(&st[t],      ssum[0][t] + ssum[1][t] + ssum[2][t] + ssum[3][t]);
    atomicAdd(&st[64 + t], ssq[0][t]  + ssq[1][t]  + ssq[2][t]  + ssq[3][t]);
  }
}

// out = (resid?) + relu?( (x-mean)*rsqrt(var+eps)*g + b )
__global__ void bn_apply_k(const float* __restrict__ X,
                           const float* __restrict__ st,
                           const float* __restrict__ gam,
                           const float* __restrict__ bet,
                           const float* __restrict__ resid,
                           float* __restrict__ out, int M, int relu) {
  size_t i = (size_t)blockIdx.x * blockDim.x + threadIdx.x;
  if (i >= (size_t)M * 64) return;
  int c = (int)(i & 63);
  float inv = 1.0f / (float)M;
  float mean = st[c] * inv;
  float var  = st[64 + c] * inv - mean * mean;
  float v = (X[i] - mean) * rsqrtf(var + 1e-5f) * gam[c] + bet[c];
  if (relu) v = fmaxf(v, 0.f);
  if (resid) v += resid[i];
  out[i] = v;
}

// pack fp32 weight W[K,64] into WMMA B-fragment layout (f16), one dword
// (half2) per thread:  out[((kc*4+nt)*32 + lane)*8 + v] = {W[k0,n], W[k0+1,n]}
// with k0 = kc*32 + (lane>>4)*16 + 2v, n = nt*16 + (lane&15)
__global__ void pack_w_k(const float* __restrict__ W, int K,
                         unsigned* __restrict__ out) {
  int tid = blockIdx.x * blockDim.x + threadIdx.x;
  int total = K * 32;
  if (tid >= total) return;
  int v  = tid & 7;
  int l  = (tid >> 3) & 31;
  int nt = (tid >> 8) & 3;
  int kc = tid >> 10;
  int k0 = kc * 32 + (l >> 4) * 16 + 2 * v;
  int n  = nt * 16 + (l & 15);
  union { _Float16 h[2]; unsigned u; } p;
  p.h[0] = (_Float16)W[(size_t)k0 * 64 + n];
  p.h[1] = (_Float16)W[(size_t)(k0 + 1) * 64 + n];
  out[tid] = p.u;
}

// final head: d_out[i] = h[i,:] @ W[:,0] + b
__global__ void final_out_k(const float* __restrict__ h,
                            const float* __restrict__ W,
                            const float* __restrict__ b,
                            float* __restrict__ out, int Bn) {
  int i = blockIdx.x * blockDim.x + threadIdx.x;
  if (i >= Bn) return;
  const float* hp = h + (size_t)i * 64;
  float a = b[0];
#pragma unroll
  for (int k = 0; k < 64; ++k) a += hp[k] * W[k];
  out[i] = a;
}

// ---------------------------------------------------------------------------
// WMMA GEMM: Y[M,64] = epi( act( A[M,K] @ W[K,64] + bias ) )
//   A-source modes: plain / concat(gather,plain) / concat(plain,plain) /
//                   0.5*(gather+gather)   (the a2e pair-mean)
//   optional per-row 1/max(cnt,1) scale (fused segment-mean finalize)
//   epilogues: coalesced store (+optional residual) or atomic scatter-add
// one wave = 16 rows x 64 cols, 4 waves / block
// ---------------------------------------------------------------------------
constexpr int A_PLAIN = 0, A_CGP = 1, A_CPP = 2, A_PAIR = 3;
constexpr int E_STORE = 0, E_SCAT = 1;

template <int K, int AMODE, int EPI>
__global__ void __launch_bounds__(128) gemm_k(
    const float* __restrict__ Xa, const int* __restrict__ idxA,
    const float* __restrict__ Xb, const int* __restrict__ idxB,
    const float* __restrict__ rowcnt,
    const unsigned* __restrict__ Wp, const float* __restrict__ bias,
    float* __restrict__ Y, const float* __restrict__ resid,
    const int* __restrict__ didx, float* __restrict__ cnt,
    int M, int relu) {
  __shared__ _Float16 sA[4][16][136];  // 272B row stride: conflict-free b128
  __shared__ float    sD[4][16][68];   // 272B row stride, 16B-aligned float4

  const int t = threadIdx.x;
  const int w = t >> 5;        // wave in block
  const int l = t & 31;        // lane
  const int g = l >> 4;        // lane half-group
  const int m0 = blockIdx.x * 64 + w * 16;

  // ---- stage A tile (fp32 -> fp16) into LDS, fusing gather / concat / mean
  constexpr int F4 = K / 4;                // float4 per row
  constexpr int ITERS = (16 * F4) / 32;    // float4 per lane
#pragma unroll
  for (int it = 0; it < ITERS; ++it) {
    int flat = it * 32 + l;
    int row = flat / F4;
    int c4 = flat % F4;
    int col = c4 * 4;
    int mrow = m0 + row;
    float v0 = 0.f, v1 = 0.f, v2 = 0.f, v3 = 0.f;
    if (mrow < M) {
      if (AMODE == A_PLAIN) {
        const float4 s = *(const float4*)(Xa + (size_t)mrow * K + col);
        v0 = s.x; v1 = s.y; v2 = s.z; v3 = s.w;
        if (rowcnt) {
          float sc = 1.0f / fmaxf(rowcnt[mrow], 1.0f);
          v0 *= sc; v1 *= sc; v2 *= sc; v3 *= sc;
        }
      } else if (AMODE == A_CGP) {  // [gather(Xa,idxA) | Xb]
        if (col < 64) {
          const float4 s = *(const float4*)(Xa + (size_t)idxA[mrow] * 64 + col);
          v0 = s.x; v1 = s.y; v2 = s.z; v3 = s.w;
        } else {
          const float4 s = *(const float4*)(Xb + (size_t)mrow * 64 + (col - 64));
          v0 = s.x; v1 = s.y; v2 = s.z; v3 = s.w;
        }
      } else if (AMODE == A_CPP) {  // [Xa | Xb]
        const float4 s = (col < 64)
            ? *(const float4*)(Xa + (size_t)mrow * 64 + col)
            : *(const float4*)(Xb + (size_t)mrow * 64 + (col - 64));
        v0 = s.x; v1 = s.y; v2 = s.z; v3 = s.w;
      } else {  // A_PAIR: 0.5*(Xa[idxA] + Xa[idxB])
        const float4 s0 = *(const float4*)(Xa + (size_t)idxA[mrow] * 64 + col);
        const float4 s1 = *(const float4*)(Xa + (size_t)idxB[mrow] * 64 + col);
        v0 = 0.5f * (s0.x + s1.x); v1 = 0.5f * (s0.y + s1.y);
        v2 = 0.5f * (s0.z + s1.z); v3 = 0.5f * (s0.w + s1.w);
      }
    }
    H4 hv;
    hv.v[0] = (_Float16)v0; hv.v[1] = (_Float16)v1;
    hv.v[2] = (_Float16)v2; hv.v[3] = (_Float16)v3;
    *(H4*)&sA[w][row][col] = hv;
  }
  __syncthreads();

  // ---- WMMA main loop: 4 N-tiles x (K/32) K-chunks
  union AF { uint4 u[2]; v16h h; };
  v8f z8 = {0.f, 0.f, 0.f, 0.f, 0.f, 0.f, 0.f, 0.f};
  v8f acc[4] = {z8, z8, z8, z8};
  const uint4* wp4 = (const uint4*)Wp;
#pragma unroll
  for (int kc = 0; kc < K / 32; ++kc) {
    AF a;
    a.u[0] = *(const uint4*)&sA[w][l & 15][kc * 32 + g * 16];
    a.u[1] = *(const uint4*)&sA[w][l & 15][kc * 32 + g * 16 + 8];
#pragma unroll
    for (int nt = 0; nt < 4; ++nt) {
      AF b;
      int base = ((kc * 4 + nt) * 32 + l) * 2;
      b.u[0] = wp4[base];
      b.u[1] = wp4[base + 1];
      acc[nt] = __builtin_amdgcn_wmma_f32_16x16x32_f16(
          false, a.h, false, b.h, (short)0, acc[nt], false, false);
    }
  }

  float bb[4];
#pragma unroll
  for (int nt = 0; nt < 4; ++nt) bb[nt] = bias[nt * 16 + (l & 15)];

  if (EPI == E_STORE) {
#pragma unroll
    for (int nt = 0; nt < 4; ++nt)
#pragma unroll
      for (int r = 0; r < 8; ++r) {
        float v = acc[nt][r] + bb[nt];
        if (relu) v = fmaxf(v, 0.f);
        sD[w][r + 8 * g][nt * 16 + (l & 15)] = v;
      }
    __syncthreads();
#pragma unroll
    for (int it = 0; it < 8; ++it) {
      int row = it * 2 + g;
      int mrow = m0 + row;
      if (mrow < M) {
        int col = (l & 15) * 4;
        float4 o = *(const float4*)&sD[w][row][col];
        if (resid) {
          float4 rr = *(const float4*)(resid + (size_t)mrow * 64 + col);
          o.x += rr.x; o.y += rr.y; o.z += rr.z; o.w += rr.w;
        }
        *(float4*)(Y + (size_t)mrow * 64 + col) = o;
      }
    }
  } else {  // E_SCAT: segment-sum directly from accumulators
#pragma unroll
    for (int r = 0; r < 8; ++r) {
      int mrow = m0 + r + 8 * g;
      if (mrow < M) {
        int d = didx[mrow];
        float* yp = Y + (size_t)d * 64 + (l & 15);
#pragma unroll
        for (int nt = 0; nt < 4; ++nt) {
          float v = acc[nt][r] + bb[nt];
          if (relu) v = fmaxf(v, 0.f);
          atomicAdd(yp + nt * 16, v);
        }
        if ((l & 15) == 0) atomicAdd(&cnt[d], 1.0f);
      }
    }
  }
}

// ---------------------------------------------------------------------------
// host orchestration
// ---------------------------------------------------------------------------
extern "C" void kernel_launch(void* const* d_in, const int* in_sizes, int n_in,
                              void* d_out, int out_size, void* d_ws,
                              size_t ws_size, hipStream_t stream) {
  const int N   = in_sizes[0] / 3;   // atoms
  const int E   = in_sizes[3] / 2;   // edges
  const int F   = in_sizes[2] / 32;  // fragments
  const int NAF = in_sizes[4] / 2;   // atom-frag links
  const int NFE = in_sizes[5] / 2;   // frag-frag links
  const int Bn  = out_size;          // graphs

  auto Pf = [&](int rel) { return (const float*)d_in[9 + rel]; };
  const int* feats  = (const int*)d_in[0];
  const int* eattr  = (const int*)d_in[1];
  const float* frin = (const float*)d_in[2];
  const int* re = (const int*)d_in[3];       const int* ce = re + E;
  const int* afA = (const int*)d_in[4];      const int* afF = afA + NAF;
  const int* fr = (const int*)d_in[5];       const int* fc = fr + NFE;
  const int* batch  = (const int*)d_in[6];
  const int* fbatch = (const int*)d_in[7];

  // params pytree (sorted keys) relative indices from d_in[9]:
  // atom_emb 0-2 | atom_out 3-10 | bond_emb 11-13 | edge_out 14-21 |
  // frag_enc 22,23 | frag_out 24-31 | layers 32..111 (40/layer) | out 112,113

  // ---- workspace partition
  char* wsb = (char*)d_ws;
  size_t off = 0;
  auto FAL = [&](size_t nf) {
    float* p = (float*)(wsb + off);
    off += CDIV(nf * 4, (size_t)256) * 256;
    return p;
  };
  float* x    = FAL((size_t)N * 64);
  float* xe   = FAL((size_t)E * 64);
  float* xf   = FAL((size_t)F * 64);
  float* accN = FAL((size_t)N * 64);
  float* cntN = FAL(N);
  float* bufN = FAL((size_t)N * 64);  // m1
  float* m2   = FAL((size_t)N * 64);
  float* bufE = FAL((size_t)E * 64);
  float* accF = FAL((size_t)F * 64);
  float* cntF = FAL(F);
  float* f1b  = FAL((size_t)F * 64);
  float* f2b  = FAL((size_t)F * 64);
  float* accB1 = FAL((size_t)Bn * 64); float* cntB1 = FAL(Bn);
  float* accB2 = FAL((size_t)Bn * 64); float* cntB2 = FAL(Bn);
  float* accB3 = FAL((size_t)Bn * 64); float* cntB3 = FAL(Bn);
  float* tB    = FAL((size_t)Bn * 64);
  float* hB    = FAL((size_t)Bn * 64);
  float* stats = FAL(128);
  auto WAL = [&](int K) {
    unsigned* p = (unsigned*)(wsb + off);
    off += CDIV((size_t)K * 32 * 4, (size_t)256) * 256;
    return p;
  };

  auto pack = [&](int rel, int K, unsigned* dst) {
    int n = K * 32;
    pack_w_k<<<CDIV(n, 256), 256, 0, stream>>>(Pf(rel), K, dst);
  };

  // ---- pack all weights into WMMA fragment layout (fp16)
  unsigned* wp_fe = WAL(32);  pack(22, 32, wp_fe);
  struct LW {
    unsigned *a2aA1, *a2aA2, *a2aB, *a2e1, *a2e2, *a2f1, *a2f2,
             *combA, *combE, *combF, *f2a1, *f2a2, *f2f1, *f2f2;
  } lw[2];
  for (int L = 0; L < 2; ++L) {
    int bs = 32 + 40 * L;
    lw[L].a2aA1 = WAL(64);  pack(bs + 0, 64, lw[L].a2aA1);
    lw[L].a2aA2 = WAL(64);  pack(bs + 2, 64, lw[L].a2aA2);
    lw[L].a2aB  = WAL(128); pack(bs + 4, 128, lw[L].a2aB);
    lw[L].a2e1  = WAL(64);  pack(bs + 6, 64, lw[L].a2e1);
    lw[L].a2e2  = WAL(64);  pack(bs + 8, 64, lw[L].a2e2);
    lw[L].a2f1  = WAL(64);  pack(bs + 10, 64, lw[L].a2f1);
    lw[L].a2f2  = WAL(64);  pack(bs + 12, 64, lw[L].a2f2);
    lw[L].combA = WAL(128); pack(bs + 20, 128, lw[L].combA);
    lw[L].combE = WAL(64);  pack(bs + 24, 64, lw[L].combE);
    lw[L].combF = WAL(128); pack(bs + 28, 128, lw[L].combF);
    lw[L].f2a1  = WAL(64);  pack(bs + 32, 64, lw[L].f2a1);
    lw[L].f2a2  = WAL(64);  pack(bs + 34, 64, lw[L].f2a2);
    lw[L].f2f1  = WAL(64);  pack(bs + 36, 64, lw[L].f2f1);
    lw[L].f2f2  = WAL(64);  pack(bs + 38, 64, lw[L].f2f2);
  }
  unsigned* wp_ao1 = WAL(64); pack(3, 64, wp_ao1);
  unsigned* wp_ao2 = WAL(64); pack(7, 64, wp_ao2);
  unsigned* wp_eo1 = WAL(64); pack(14, 64, wp_eo1);
  unsigned* wp_eo2 = WAL(64); pack(18, 64, wp_eo2);
  unsigned* wp_fo1 = WAL(64); pack(24, 64, wp_fo1);
  unsigned* wp_fo2 = WAL(64); pack(28, 64, wp_fo2);

  auto zeroF = [&](float* p, size_t n) {
    zero_k<<<CDIV(n, (size_t)256), 256, 0, stream>>>(p, n);
  };
  auto stat = [&](const float* X, int M) {
    zeroF(stats, 128);
    int nb = (M + 3) / 4; if (nb > 2048) nb = 2048; if (nb < 1) nb = 1;
    colstats_k<<<nb, 256, 0, stream>>>(X, M, stats);
  };
  auto bn = [&](const float* X, const float* gg, const float* bb,
                const float* resid, float* out, int M, int relu) {
    bn_apply_k<<<CDIV((size_t)M * 64, (size_t)256), 256, 0, stream>>>(
        X, stats, gg, bb, resid, out, M, relu);
  };
  // plain [M,64]@[64,64] GEMM helper
  auto g64 = [&](const float* A, const float* rc, const unsigned* W,
                 const float* bias, float* Y, const float* resid, int M,
                 int relu) {
    gemm_k<64, A_PLAIN, E_STORE><<<CDIV(M, 64), 128, 0, stream>>>(
        A, nullptr, nullptr, nullptr, rc, W, bias, Y, resid, nullptr, nullptr,
        M, relu);
  };

  // ---- encoders
  encode3_k<<<CDIV((size_t)N * 64, (size_t)256), 256, 0, stream>>>(
      feats, Pf(0), Pf(1), Pf(2), x, N);
  encode3_k<<<CDIV((size_t)E * 64, (size_t)256), 256, 0, stream>>>(
      eattr, Pf(11), Pf(12), Pf(13), xe, E);
  gemm_k<32, A_PLAIN, E_STORE><<<CDIV(F, 64), 128, 0, stream>>>(
      frin, nullptr, nullptr, nullptr, nullptr, wp_fe, Pf(23), xf, nullptr,
      nullptr, nullptr, F, 0);

  // ---- message-passing layers
  for (int L = 0; L < 2; ++L) {
    int bs = 32 + 40 * L;
    // a2a: relu([x[re]|xe] @ Wb + b) -> seg-mean over ce (fused scatter)
    zeroF(accN, (size_t)N * 64); zeroF(cntN, N);
    gemm_k<128, A_CGP, E_SCAT><<<CDIV(E, 64), 128, 0, stream>>>(
        x, re, xe, nullptr, nullptr, lw[L].a2aB, Pf(bs + 5), accN, nullptr, ce,
        cntN, E, 1);
    g64(accN, cntN, lw[L].a2aA1, Pf(bs + 1), bufN, nullptr, N, 1);
    g64(bufN, nullptr, lw[L].a2aA2, Pf(bs + 3), bufN, x, N, 1);  // m1 = x + ...
    // f2a: seg-mean(xf[col] -> row) -> MLP
    zeroF(accN, (size_t)N * 64); zeroF(cntN, N);
    scatter_rows_k<<<CDIV((size_t)NAF * 64, (size_t)256), 256, 0, stream>>>(
        xf, afF, afA, nullptr, accN, cntN, NAF);
    g64(accN, cntN, lw[L].f2a1, Pf(bs + 33), m2, nullptr, N, 1);
    g64(m2, nullptr, lw[L].f2a2, Pf(bs + 35), m2, nullptr, N, 1);
    // comb_atom: [m1|m2] @ Wc + b -> BN+ReLU, residual, then bn_atom
    gemm_k<128, A_CPP, E_STORE><<<CDIV(N, 64), 128, 0, stream>>>(
        bufN, nullptr, m2, nullptr, nullptr, lw[L].combA, Pf(bs + 21), m2,
        nullptr, nullptr, nullptr, N, 0);
    stat(m2, N); bn(m2, Pf(bs + 22), Pf(bs + 23), x, x, N, 1);
    stat(x, N);  bn(x, Pf(bs + 14), Pf(bs + 15), nullptr, x, N, 1);
    // a2e: mean(x[re],x[ce]) -> MLP -> comb_edge -> BN chain
    gemm_k<64, A_PAIR, E_STORE><<<CDIV(E, 64), 128, 0, stream>>>(
        x, re, nullptr, ce, nullptr, lw[L].a2e1, Pf(bs + 7), bufE, nullptr,
        nullptr, nullptr, E, 1);
    g64(bufE, nullptr, lw[L].a2e2, Pf(bs + 9), bufE, nullptr, E, 1);
    g64(bufE, nullptr, lw[L].combE, Pf(bs + 25), bufE, nullptr, E, 0);
    stat(bufE, E); bn(bufE, Pf(bs + 26), Pf(bs + 27), xe, xe, E, 1);
    stat(xe, E);   bn(xe, Pf(bs + 16), Pf(bs + 17), nullptr, xe, E, 1);
    // a2f: seg-mean(x[row] -> col) -> MLP
    zeroF(accF, (size_t)F * 64); zeroF(cntF, F);
    scatter_rows_k<<<CDIV((size_t)NAF * 64, (size_t)256), 256, 0, stream>>>(
        x, afA, afF, nullptr, accF, cntF, NAF);
    g64(accF, cntF, lw[L].a2f1, Pf(bs + 11), f1b, nullptr, F, 1);
    g64(f1b, nullptr, lw[L].a2f2, Pf(bs + 13), f1b, nullptr, F, 1);
    // f2f: seg-mean(xf[fr] -> fc) -> MLP
    zeroF(accF, (size_t)F * 64); zeroF(cntF, F);
    scatter_rows_k<<<CDIV((size_t)NFE * 64, (size_t)256), 256, 0, stream>>>(
        xf, fr, fc, nullptr, accF, cntF, NFE);
    g64(accF, cntF, lw[L].f2f1, Pf(bs + 37), f2b, nullptr, F, 1);
    g64(f2b, nullptr, lw[L].f2f2, Pf(bs + 39), f2b, nullptr, F, 1);
    // comb_frag + BN chain
    gemm_k<128, A_CPP, E_STORE><<<CDIV(F, 64), 128, 0, stream>>>(
        f1b, nullptr, f2b, nullptr, nullptr, lw[L].combF, Pf(bs + 29), f2b,
        nullptr, nullptr, nullptr, F, 0);
    stat(f2b, F); bn(f2b, Pf(bs + 30), Pf(bs + 31), xf, xf, F, 1);
    stat(xf, F);  bn(xf, Pf(bs + 18), Pf(bs + 19), nullptr, xf, F, 1);
  }

  // ---- readout: per-graph means of atoms / frags / edges
  zeroF(accB1, (size_t)Bn * 64); zeroF(cntB1, Bn);
  zeroF(accB2, (size_t)Bn * 64); zeroF(cntB2, Bn);
  zeroF(accB3, (size_t)Bn * 64); zeroF(cntB3, Bn);
  zeroF(hB, (size_t)Bn * 64);
  scatter_rows_k<<<CDIV((size_t)N * 64, (size_t)256), 256, 0, stream>>>(
      x, nullptr, batch, nullptr, accB1, cntB1, N);
  scatter_rows_k<<<CDIV((size_t)F * 64, (size_t)256), 256, 0, stream>>>(
      xf, nullptr, fbatch, nullptr, accB2, cntB2, F);
  scatter_rows_k<<<CDIV((size_t)E * 64, (size_t)256), 256, 0, stream>>>(
      xe, nullptr, re, batch, accB3, cntB3, E);

  // three 2-layer Linear+BN+ReLU branches accumulated into hB
  auto branch = [&](float* acc, float* cntv, const unsigned* W1, int r1,
                    const unsigned* W2, int r2) {
    g64(acc, cntv, W1, Pf(r1 + 1), tB, nullptr, Bn, 0);
    stat(tB, Bn); bn(tB, Pf(r1 + 2), Pf(r1 + 3), nullptr, tB, Bn, 1);
    g64(tB, nullptr, W2, Pf(r2 + 1), tB, nullptr, Bn, 0);
    stat(tB, Bn); bn(tB, Pf(r2 + 2), Pf(r2 + 3), hB, hB, Bn, 1);
  };
  branch(accB1, cntB1, wp_ao1, 3, wp_ao2, 7);    // atom_out
  branch(accB2, cntB2, wp_fo1, 24, wp_fo2, 28);  // frag_out
  branch(accB3, cntB3, wp_eo1, 14, wp_eo2, 18);  // edge_out

  final_out_k<<<CDIV(Bn, 256), 256, 0, stream>>>(hB, Pf(112), Pf(113),
                                                 (float*)d_out, Bn);
}
